// TopoDPConv_4750233829391
// MI455X (gfx1250) — matrix operand order repaired and checked by reading, not compiled, not run on hardware
//
#include <hip/hip_runtime.h>

typedef _Float16 v16h __attribute__((ext_vector_type(16)));
typedef _Float16 v8h  __attribute__((ext_vector_type(8)));
typedef float    v8f  __attribute__((ext_vector_type(8)));
typedef float    v4f  __attribute__((ext_vector_type(4)));

#define HID 128
#define NLEV 4

// ---------------------------------------------------------------------------
// A-fragment load from an LDS fp16 tile (row-major, `stride` halves per row).
// 16x32 f16 A layout (wave32): lane m=lane%16 holds row m; half-wave hs=lane/16
// elems[0..7]  = K = kb + hs*8 + 0..7
// elems[8..15] = K = kb + 16 + hs*8 + 0..7
// ---------------------------------------------------------------------------
__device__ __forceinline__ v16h ldA(const _Float16* t, int stride, int m, int hs, int kb) {
    const _Float16* p = t + m * stride + kb + hs * 8;
    v8h lo = *(const v8h*)p;
    v8h hi = *(const v8h*)(p + 16);
    v16h a;
#pragma unroll
    for (int i = 0; i < 8; ++i) { a[i] = lo[i]; a[i + 8] = hi[i]; }
    return a;
}

// ---------------------------------------------------------------------------
// Wave-level GEMM: [16 x (NKC*32)] (LDS, f16) x [(NKC*32) x 128] (global,
// fragment-ordered f16 weights) -> 8 accumulators of 16x16 f32.
// ---------------------------------------------------------------------------
template <int NKC>
__device__ __forceinline__ void wave_gemm(const _Float16* at, int stride,
                                          const _Float16* wf, v8f* acc, int lane) {
    int m = lane & 15, hs = lane >> 4;
#pragma unroll
    for (int kc = 0; kc < NKC; ++kc) {
        v16h a = ldA(at, stride, m, hs, kc * 32);
#pragma unroll
        for (int ct = 0; ct < 8; ++ct) {
            v16h b = *(const v16h*)(wf + (size_t)(((ct * NKC) + kc) * 32 + lane) * 16);
            acc[ct] = __builtin_amdgcn_wmma_f32_16x16x32_f16(
                false, a, false, b, (short)0, acc[ct], false, false);
        }
    }
}

// ---------------------------------------------------------------------------
// Convert a KxHID fp32 weight matrix to fp16, stored in per-lane B-fragment
// order: F[((ct*nK + kc)*32 + lane)*16 + e] = W[kc*32 + (lane/16)*8 + (e%8) + (e/8)*16][ct*16 + lane%16]
// ---------------------------------------------------------------------------
__global__ void k_wfrag(const float* __restrict__ W, _Float16* __restrict__ F, int K) {
    int nK = K >> 5;
    int total = K * HID;
    int tid = blockIdx.x * 256 + threadIdx.x;
    if (tid >= total) return;
    int e    = tid & 15;
    int lane = (tid >> 4) & 31;
    int grp  = tid >> 9;           // ct*nK + kc
    int kc   = grp % nK;
    int ct   = grp / nK;
    int k = kc * 32 + (lane >> 4) * 8 + (e & 7) + (e >> 3) * 16;
    int n = ct * 16 + (lane & 15);
    F[tid] = (_Float16)W[(size_t)k * HID + n];
}

// ---------------------------------------------------------------------------
// Bucket edges by depth[dst] and nodes by depth into per-level compact lists.
// ---------------------------------------------------------------------------
__global__ void k_lists(const int* __restrict__ src_unused, const int* __restrict__ dst,
                        const int* __restrict__ depth,
                        int* __restrict__ elist, int* __restrict__ ecnt,
                        int* __restrict__ nlist, int* __restrict__ ncnt, int E, int N) {
    int i = blockIdx.x * 256 + threadIdx.x;
    if (i < E) {
        int d = depth[dst[i]];
        if (d >= 1 && d <= NLEV) {
            int p = atomicAdd(&ecnt[d - 1], 1);
            elist[(size_t)(d - 1) * E + p] = i;
        }
    }
    if (i < N) {
        int d = depth[i];
        if (d >= 1 && d <= NLEV) {
            int p = atomicAdd(&ncnt[d - 1], 1);
            nlist[(size_t)(d - 1) * N + p] = i;
        }
    }
}

// ---------------------------------------------------------------------------
// Projection: h = x @ pw + pb  (fp32 out + fp16 mirror). One wave per 16 rows.
// ---------------------------------------------------------------------------
__global__ void k_proj(const float* __restrict__ x, const _Float16* __restrict__ pwf,
                       const float* __restrict__ pb, float* __restrict__ h,
                       _Float16* __restrict__ h16, int N) {
    __shared__ _Float16 sA[4][16 * 136];
    int lane = threadIdx.x & 31, wid = threadIdx.x >> 5;
    int tile = blockIdx.x * 4 + wid;
    if (tile * 16 >= N) return;
    _Float16* at = sA[wid];
#pragma unroll
    for (int it = 0; it < 16; ++it) {
        int idx = it * 32 + lane;             // 0..511
        int r = idx >> 5, seg = idx & 31;
        int row = tile * 16 + r; if (row >= N) row = N - 1;
        v4f v = *(const v4f*)(x + (size_t)row * HID + seg * 4);
        _Float16* dp = at + r * 136 + seg * 4;
        dp[0] = (_Float16)v[0]; dp[1] = (_Float16)v[1];
        dp[2] = (_Float16)v[2]; dp[3] = (_Float16)v[3];
    }
    v8f acc[8] = {};
    wave_gemm<4>(at, 136, pwf, acc, lane);
    int m = lane & 15, hs = lane >> 4;
#pragma unroll
    for (int ct = 0; ct < 8; ++ct) {
        int ncol = ct * 16 + m;
        float b = pb[ncol];
#pragma unroll
        for (int r = 0; r < 8; ++r) {
            int row = tile * 16 + r + hs * 8;
            if (row < N) {
                float val = acc[ct][r] + b;
                h[(size_t)row * HID + ncol]   = val;
                h16[(size_t)row * HID + ncol] = (_Float16)val;
            }
        }
    }
}

// ---------------------------------------------------------------------------
// Edge message MLP for one level: for 16-edge tiles from the level's list,
// gather h16[src], 2-layer MLP (129->128->128 with relu), float-atomic scatter
// into agg[dst], and bump counts[dst].
// ---------------------------------------------------------------------------
__global__ void k_edge(const _Float16* __restrict__ h16, const int* __restrict__ src,
                       const int* __restrict__ dst, const float* __restrict__ ea,
                       const _Float16* __restrict__ w1f, const float* __restrict__ w1last,
                       const float* __restrict__ b1, const _Float16* __restrict__ w2f,
                       const float* __restrict__ b2, const int* __restrict__ list,
                       const int* __restrict__ cnts, int lvl,
                       float* __restrict__ agg, float* __restrict__ counts) {
    __shared__ _Float16 sA[4][16 * 136];
    __shared__ _Float16 sM[4][16 * 136];
    __shared__ int   sSrc[4][16];
    __shared__ int   sDst[4][16];
    __shared__ float sEa[4][16];
    int lane = threadIdx.x & 31, wid = threadIdx.x >> 5;
    int cnt = cnts[lvl];
    int tile = blockIdx.x * 4 + wid;
    if (tile * 16 >= cnt) return;
    int nrows = cnt - tile * 16; if (nrows > 16) nrows = 16;
    if (lane < 16) {
        int e;
        if (lane < nrows) {
            e = list[tile * 16 + lane];
            atomicAdd(&counts[dst[e]], 1.0f);
            sEa[wid][lane]  = ea[e];
            sDst[wid][lane] = dst[e];
        } else {
            e = list[tile * 16];
            sEa[wid][lane]  = 0.0f;
            sDst[wid][lane] = 0;
        }
        sSrc[wid][lane] = src[e];
    }
    _Float16* at = sA[wid];
#pragma unroll
    for (int it = 0; it < 8; ++it) {
        int idx = it * 32 + lane;             // 0..255
        int r = idx >> 4, seg = idx & 15;
        *(v8h*)(at + r * 136 + seg * 8) =
            *(const v8h*)(h16 + (size_t)sSrc[wid][r] * HID + seg * 8);
    }
    v8f acc[8] = {};
    wave_gemm<4>(at, 136, w1f, acc, lane);
    int m = lane & 15, hs = lane >> 4;
    _Float16* mt = sM[wid];
#pragma unroll
    for (int ct = 0; ct < 8; ++ct) {
        int ncol = ct * 16 + m;
        float wl = w1last[ncol], bb = b1[ncol];
#pragma unroll
        for (int r = 0; r < 8; ++r) {
            int mr = r + hs * 8;
            float val = acc[ct][r] + sEa[wid][mr] * wl + bb;
            val = val > 0.0f ? val : 0.0f;
            mt[mr * 136 + ncol] = (_Float16)val;
        }
    }
    v8f acc2[8] = {};
    wave_gemm<4>(mt, 136, w2f, acc2, lane);
#pragma unroll
    for (int ct = 0; ct < 8; ++ct) {
        int ncol = ct * 16 + m;
        float bb = b2[ncol];
#pragma unroll
        for (int r = 0; r < 8; ++r) {
            int mr = r + hs * 8;
            if (mr < nrows) {
                float val = acc2[ct][r] + bb;
                val = val > 0.0f ? val : 0.0f;
                atomicAdd(&agg[(size_t)sDst[wid][mr] * HID + ncol], val);
            }
        }
    }
}

// ---------------------------------------------------------------------------
// Node update MLP for one level: gather [h16 ; agg/clamp(counts,1)] (K=256),
// 2-layer MLP with relu, write back h (fp32) and h16 only for listed nodes.
// ---------------------------------------------------------------------------
__global__ void k_node(float* __restrict__ h, _Float16* __restrict__ h16,
                       const float* __restrict__ agg, const float* __restrict__ counts,
                       const _Float16* __restrict__ u1f, const float* __restrict__ b1,
                       const _Float16* __restrict__ u2f, const float* __restrict__ b2,
                       const int* __restrict__ list, const int* __restrict__ cnts, int lvl) {
    __shared__ _Float16 sA[4][16 * 264];
    __shared__ _Float16 sM[4][16 * 136];
    __shared__ int   sId[4][16];
    __shared__ float sDen[4][16];
    int lane = threadIdx.x & 31, wid = threadIdx.x >> 5;
    int cnt = cnts[lvl];
    int tile = blockIdx.x * 4 + wid;
    if (tile * 16 >= cnt) return;
    int nrows = cnt - tile * 16; if (nrows > 16) nrows = 16;
    if (lane < 16) {
        int id = list[tile * 16 + (lane < nrows ? lane : 0)];
        sId[wid][lane] = id;
        float c = counts[id];
        sDen[wid][lane] = 1.0f / (c > 1.0f ? c : 1.0f);
    }
    _Float16* at = sA[wid];
#pragma unroll
    for (int it = 0; it < 8; ++it) {
        int idx = it * 32 + lane;
        int r = idx >> 4, seg = idx & 15;
        *(v8h*)(at + r * 264 + seg * 8) =
            *(const v8h*)(h16 + (size_t)sId[wid][r] * HID + seg * 8);
    }
#pragma unroll
    for (int it = 0; it < 16; ++it) {
        int idx = it * 32 + lane;             // 0..511
        int r = idx >> 5, seg = idx & 31;
        v4f v = *(const v4f*)(agg + (size_t)sId[wid][r] * HID + seg * 4);
        float dn = sDen[wid][r];
        _Float16* dp = at + r * 264 + 128 + seg * 4;
        dp[0] = (_Float16)(v[0] * dn); dp[1] = (_Float16)(v[1] * dn);
        dp[2] = (_Float16)(v[2] * dn); dp[3] = (_Float16)(v[3] * dn);
    }
    v8f acc[8] = {};
    wave_gemm<8>(at, 264, u1f, acc, lane);
    int m = lane & 15, hs = lane >> 4;
    _Float16* mt = sM[wid];
#pragma unroll
    for (int ct = 0; ct < 8; ++ct) {
        int ncol = ct * 16 + m;
        float bb = b1[ncol];
#pragma unroll
        for (int r = 0; r < 8; ++r) {
            int mr = r + hs * 8;
            float val = acc[ct][r] + bb;
            val = val > 0.0f ? val : 0.0f;
            mt[mr * 136 + ncol] = (_Float16)val;
        }
    }
    v8f acc2[8] = {};
    wave_gemm<4>(mt, 136, u2f, acc2, lane);
#pragma unroll
    for (int ct = 0; ct < 8; ++ct) {
        int ncol = ct * 16 + m;
        float bb = b2[ncol];
#pragma unroll
        for (int r = 0; r < 8; ++r) {
            int mr = r + hs * 8;
            if (mr < nrows) {
                int id = sId[wid][mr];
                float val = acc2[ct][r] + bb;
                val = val > 0.0f ? val : 0.0f;
                h[(size_t)id * HID + ncol]   = val;
                h16[(size_t)id * HID + ncol] = (_Float16)val;
            }
        }
    }
}

// ---------------------------------------------------------------------------
extern "C" void kernel_launch(void* const* d_in, const int* in_sizes, int n_in,
                              void* d_out, int out_size, void* d_ws, size_t ws_size,
                              hipStream_t stream) {
    const float* x     = (const float*)d_in[0];
    const int*   eidx  = (const int*)  d_in[1];
    const float* eattr = (const float*)d_in[2];
    const int*   depth = (const int*)  d_in[3];
    const float* pw  = (const float*)d_in[4];
    const float* pb  = (const float*)d_in[5];
    const float* mw1 = (const float*)d_in[6];
    const float* mb1 = (const float*)d_in[7];
    const float* mw2 = (const float*)d_in[8];
    const float* mb2 = (const float*)d_in[9];
    const float* uw1 = (const float*)d_in[10];
    const float* ub1 = (const float*)d_in[11];
    const float* uw2 = (const float*)d_in[12];
    const float* ub2 = (const float*)d_in[13];

    int N = in_sizes[0] / HID;
    int E = in_sizes[1] / 2;
    const int* src = eidx;
    const int* dst = eidx + E;
    float* h = (float*)d_out;

    // Carve workspace (256B-aligned slabs).
    char* wsp = (char*)d_ws;
    size_t off = 0;
    auto take = [&](size_t bytes) -> char* {
        char* p = wsp + off;
        off += (bytes + 255) & ~(size_t)255;
        return p;
    };
    _Float16* h16    = (_Float16*)take((size_t)N * HID * 2);
    float*    agg    = (float*)   take((size_t)N * HID * 4);
    float*    counts = (float*)   take((size_t)N * 4);
    int*      ecnt   = (int*)     take(64);
    int*      ncnt   = (int*)     take(64);
    int*      elist  = (int*)     take((size_t)NLEV * E * 4);
    int*      nlist  = (int*)     take((size_t)NLEV * N * 4);
    _Float16* pwf    = (_Float16*)take((size_t)HID * HID * 2);
    _Float16* w1f    = (_Float16*)take((size_t)HID * HID * 2);
    _Float16* w2f    = (_Float16*)take((size_t)HID * HID * 2);
    _Float16* u1f    = (_Float16*)take((size_t)256 * HID * 2);
    _Float16* u2f    = (_Float16*)take((size_t)HID * HID * 2);
    const float* w1last = mw1 + (size_t)HID * HID;  // 129th input row of mw1

    hipMemsetAsync(ecnt, 0, 64, stream);
    hipMemsetAsync(ncnt, 0, 64, stream);

    k_wfrag<<<(HID * HID + 255) / 256, 256, 0, stream>>>(pw,  pwf, HID);
    k_wfrag<<<(HID * HID + 255) / 256, 256, 0, stream>>>(mw1, w1f, HID);
    k_wfrag<<<(HID * HID + 255) / 256, 256, 0, stream>>>(mw2, w2f, HID);
    k_wfrag<<<(256 * HID + 255) / 256, 256, 0, stream>>>(uw1, u1f, 256);
    k_wfrag<<<(HID * HID + 255) / 256, 256, 0, stream>>>(uw2, u2f, HID);

    int lmax = E > N ? E : N;
    k_lists<<<(lmax + 255) / 256, 256, 0, stream>>>(src, dst, depth, elist, ecnt,
                                                    nlist, ncnt, E, N);

    int ptiles = (N + 15) / 16;
    k_proj<<<(ptiles + 3) / 4, 128, 0, stream>>>(x, pwf, pb, h, h16, N);

    int eblocks = (((E + 15) / 16) + 3) / 4;   // over-provisioned; waves self-trim via ecnt
    int nblocks = (((N + 15) / 16) + 3) / 4;
    for (int d = 0; d < NLEV; ++d) {
        hipMemsetAsync(agg,    0, (size_t)N * HID * 4, stream);
        hipMemsetAsync(counts, 0, (size_t)N * 4,       stream);
        k_edge<<<eblocks, 128, 0, stream>>>(h16, src, dst, eattr, w1f, w1last, mb1,
                                            w2f, mb2, elist + (size_t)d * E, ecnt, d,
                                            agg, counts);
        k_node<<<nblocks, 128, 0, stream>>>(h, h16, agg, counts, u1f, ub1, u2f, ub2,
                                            nlist + (size_t)d * N, ncnt, d);
    }
}
